// SupConLoss_26285199851606
// MI455X (gfx1250) — matrix operand order repaired and checked by reading, not compiled, not run on hardware
//
#include <hip/hip_runtime.h>
#include <hip/hip_bf16.h>

typedef __attribute__((ext_vector_type(16))) _Float16 v16h;
typedef __attribute__((ext_vector_type(8)))  float    v8f;

#define BATCH     4096
#define VIEWS     2
#define DIM       128
#define NROWS     (BATCH * VIEWS)            // 8192 anchors/contrasts
#define NTILES    (NROWS / 16)               // 512 16-wide tiles
#define COL_SPLIT 8                          // column chunks per row-tile
#define TILES_PER_CHUNK (NTILES / COL_SPLIT) // 64
#define INV_T     14.285714285714286f        // 1 / 0.07
#define EXP2_C    20.609929155556664f        // log2(e) / 0.07

// ---------------------------------------------------------------------------
// Kernel 1: L2-normalize rows and emit fp16 contrast matrix in view-major
// order: contrast[r] = features[r % B, r / B, :].  One wave per row.
// ---------------------------------------------------------------------------
__global__ __launch_bounds__(256) void normalize_kernel(
    const float* __restrict__ feat, _Float16* __restrict__ con)
{
    int wid  = (blockIdx.x * blockDim.x + threadIdx.x) >> 5;  // row id
    int lane = threadIdx.x & 31;
    if (wid >= NROWS) return;
    int b = wid & (BATCH - 1);
    int v = wid >> 12;                                        // wid / BATCH
    const float4* src = (const float4*)(feat + (size_t)(b * VIEWS + v) * DIM);
    float4 x = src[lane];
    float s = x.x * x.x + x.y * x.y + x.z * x.z + x.w * x.w;
#pragma unroll
    for (int m = 1; m < 32; m <<= 1) s += __shfl_xor(s, m, 32);
    float inv = 1.0f / sqrtf(s);
    union { _Float16 h[4]; float2 f2; } o;
    o.h[0] = (_Float16)(x.x * inv);
    o.h[1] = (_Float16)(x.y * inv);
    o.h[2] = (_Float16)(x.z * inv);
    o.h[3] = (_Float16)(x.w * inv);
    *(float2*)(con + (size_t)wid * DIM + lane * 4) = o.f2;
}

// ---------------------------------------------------------------------------
// Kernel 2: fused C*C^T (fp16 WMMA, fp32 accum) + streaming row reductions.
// Block = one 16-row tile; wave w = column chunk w (64 column tiles each).
// Per lane: n = lane%16 (column / A-row select), h = lane/16 (K-half).
// B fragments + labels are double-buffered across tiles so global loads for
// tile t+1 are in flight while tile t's 4 WMMAs + epilogue execute.
// ---------------------------------------------------------------------------
__global__ __launch_bounds__(256) void supcon_gemm_kernel(
    const _Float16* __restrict__ con, const int* __restrict__ lab,
    float* __restrict__ zP, float* __restrict__ pP, float* __restrict__ mP)
{
    const int lane     = threadIdx.x & 31;
    const int n        = lane & 15;
    const int h        = lane >> 4;
    const int rowtile  = blockIdx.x;          // 0..511
    const int colchunk = threadIdx.x >> 5;    // 0..7
    const int rowbase  = rowtile * 16;

    // A fragments: 16 rows x K=128 (contiguous K order: k = 16h + s, matches B)
    union AF { v16h v; float4 q[2]; };
    AF a[4];
    const _Float16* aRow = con + (size_t)(rowbase + n) * DIM;
#pragma unroll
    for (int kk = 0; kk < 4; ++kk) {
        a[kk].q[0] = *(const float4*)(aRow + kk * 32 + h * 16);
        a[kk].q[1] = *(const float4*)(aRow + kk * 32 + h * 16 + 8);
    }

    int rowLab[8];
#pragma unroll
    for (int g = 0; g < 8; ++g)
        rowLab[g] = lab[(rowbase + 8 * h + g) & (BATCH - 1)];

    float z[8], p[8], mc[8];
#pragma unroll
    for (int g = 0; g < 8; ++g) { z[g] = 0.f; p[g] = 0.f; mc[g] = 0.f; }

    auto loadB = [&](int t, v16h (&bb)[4], int& lb) {
        int j = (colchunk * TILES_PER_CHUNK + t) * 16 + n;
        const _Float16* bRow = con + (size_t)j * DIM;  // B column n == contrast row j
#pragma unroll
        for (int kk = 0; kk < 4; ++kk)
            bb[kk] = *(const v16h*)(bRow + kk * 32 + h * 16);  // contiguous 32B
        lb = lab[j & (BATCH - 1)];
    };

    auto computeTile = [&](int t, const v16h (&bb)[4], int colLab) {
        v8f acc = {0.f, 0.f, 0.f, 0.f, 0.f, 0.f, 0.f, 0.f};
#pragma unroll
        for (int kk = 0; kk < 4; ++kk)
            acc = __builtin_amdgcn_wmma_f32_16x16x32_f16(
                false, a[kk].v, false, bb[kk], (short)0, acc, false, false);
        int j = (colchunk * TILES_PER_CHUNK + t) * 16 + n;
#pragma unroll
        for (int g = 0; g < 8; ++g) {
            // fixed shift: row max == self-sim == 1.0 (log-softmax shift-invariant)
            float u = acc[g] - 1.0f;                  // logit * T, in [-2, 0]
            // arg in [-41.2, 0]: no subnormal guard needed -> raw v_exp_f32
            float e = __builtin_amdgcn_exp2f(u * EXP2_C);
            int  rg   = rowbase + 8 * h + g;          // global row of this value
            bool self = (j == rg);
            z[g] += self ? 0.0f : e;
            bool pos = (colLab == rowLab[g]) && !self;
            p[g]  += pos ? u : 0.0f;                  // scaled by INV_T once at end
            mc[g] += pos ? 1.0f : 0.0f;
        }
    };

    v16h b0[4], b1[4];
    int  l0, l1;
    loadB(0, b0, l0);
    for (int t = 0; t < TILES_PER_CHUNK; t += 2) {
        loadB(t + 1, b1, l1);                 // prefetch next tile
        computeTile(t, b0, l0);
        int tn = (t + 2 < TILES_PER_CHUNK) ? t + 2 : 0;  // tail: harmless reload
        loadB(tn, b0, l0);
        computeTile(t + 1, b1, l1);
    }

    // sum across the 16 lanes (distinct columns) within each half-wave
#pragma unroll
    for (int g = 0; g < 8; ++g) {
#pragma unroll
        for (int m = 1; m < 16; m <<= 1) {
            z[g]  += __shfl_xor(z[g],  m, 32);
            p[g]  += __shfl_xor(p[g],  m, 32);
            mc[g] += __shfl_xor(mc[g], m, 32);
        }
    }

    if (n == 0) {  // lanes 0 and 16 write rows 0-7 / 8-15 of the tile
#pragma unroll
        for (int g = 0; g < 8; ++g) {
            int row = rowbase + 8 * h + g;
            size_t idx = (size_t)colchunk * NROWS + row;   // disjoint slots: no atomics
            zP[idx] = z[g];
            pP[idx] = p[g] * INV_T;
            mP[idx] = mc[g];
        }
    }
}

// ---------------------------------------------------------------------------
// Kernel 3: combine column-chunk partials, per-row loss, deterministic mean.
// ---------------------------------------------------------------------------
__global__ __launch_bounds__(1024) void reduce_kernel(
    const float* __restrict__ zP, const float* __restrict__ pP,
    const float* __restrict__ mP, float* __restrict__ out)
{
    __shared__ float sdata[1024];
    int tid = threadIdx.x;
    float local = 0.f;
    for (int r = tid; r < NROWS; r += 1024) {
        float z = 0.f, p = 0.f, m = 0.f;
#pragma unroll
        for (int c = 0; c < COL_SPLIT; ++c) {
            z += zP[c * NROWS + r];
            p += pP[c * NROWS + r];
            m += mP[c * NROWS + r];
        }
        float mlpp = (p - m * __logf(z + 1e-12f)) / (m + 1e-12f);
        local += -mlpp;                       // T / BASE_T == 1
    }
    sdata[tid] = local;
    __syncthreads();
    for (int s = 512; s > 0; s >>= 1) {
        if (tid < s) sdata[tid] += sdata[tid + s];
        __syncthreads();
    }
    if (tid == 0) out[0] = sdata[0] * (1.0f / NROWS);
}

// ---------------------------------------------------------------------------
extern "C" void kernel_launch(void* const* d_in, const int* in_sizes, int n_in,
                              void* d_out, int out_size, void* d_ws, size_t ws_size,
                              hipStream_t stream)
{
    const float* feat = (const float*)d_in[0];
    const int*   lab  = (const int*)d_in[1];
    float*       out  = (float*)d_out;

    char* ws = (char*)d_ws;
    _Float16* con = (_Float16*)ws;                         // 8192*128*2  = 2 MiB
    float* zP = (float*)(ws + (size_t)NROWS * DIM * 2);    // 8*8192*4    = 256 KiB
    float* pP = zP + (size_t)COL_SPLIT * NROWS;
    float* mP = pP + (size_t)COL_SPLIT * NROWS;

    normalize_kernel<<<NROWS / 8, 256, 0, stream>>>(feat, con);
    supcon_gemm_kernel<<<NTILES, 256, 0, stream>>>(con, lab, zP, pP, mP);
    reduce_kernel<<<1, 1024, 0, stream>>>(zP, pP, mP, out);
}
